// CrissCrossAttention_54451595378930
// MI455X (gfx1250) — matrix operand order repaired and checked by reading, not compiled, not run on hardware
//
#include <hip/hip_runtime.h>
#include <hip/hip_bf16.h>
#include <cstdint>

// ---------------------------------------------------------------------------
// CDNA5 / gfx1250: wave32, WMMA 16x16x32 bf16 (f32 accumulate),
// async global->LDS DMA staging (ASYNCcnt).
// ---------------------------------------------------------------------------

typedef __attribute__((ext_vector_type(16))) __bf16 v16bf;
typedef __attribute__((ext_vector_type(8)))  float  v8f;

#define NHEADS 12
#define HD     64
#define CDIM   768
#define QKVDIM 2304

__device__ __forceinline__ float bf2f(unsigned short b) {
    unsigned int u = ((unsigned int)b) << 16;
    return __builtin_bit_cast(float, u);
}
__device__ __forceinline__ unsigned short f2bf(float f) {
    unsigned int u = __builtin_bit_cast(unsigned int, f);
    u += 0x7fffu + ((u >> 16) & 1u);   // round-to-nearest-even
    return (unsigned short)(u >> 16);
}

// CDNA5 async DMA: copy 16B per lane from global memory directly into LDS.
// VDST = LDS byte offset (workgroup-relative), VADDR = 64-bit global address.
__device__ __forceinline__ void async_copy_b128(uint32_t lds_off, uint64_t gaddr) {
    asm volatile("global_load_async_to_lds_b128 %0, %1, off"
                 :: "v"(lds_off), "v"(gaddr) : "memory");
}
__device__ __forceinline__ void wait_async0() {
    asm volatile("s_wait_asynccnt 0x0" ::: "memory");
}

// ---------------------------------------------------------------------------
// fp32 -> bf16 converter
// ---------------------------------------------------------------------------
__global__ __launch_bounds__(256)
void f32_to_bf16_kernel(const float* __restrict__ in, unsigned short* __restrict__ out, int n) {
    int i = blockIdx.x * 256 + threadIdx.x;
    if (i < n) out[i] = f2bf(in[i]);
}

// ---------------------------------------------------------------------------
// Tiled bf16 WMMA GEMM:  C[M,Nc] = A[M,K] * B[K,Nc]  (+ optional f32 bias)
// Block tile 128x128, K-step 32, 8 waves (4 along M x 2 along N),
// each wave: 2x4 grid of 16x16 f32 accumulators -> 8 v_wmma per K-step.
// Staging: double-buffered GLOBAL_LOAD_ASYNC_TO_LDS_B128 (overlaps WMMA).
// ---------------------------------------------------------------------------
union Frag { v16bf v; uint32_t u[8]; };

template<bool OUT_BF16>
__global__ __launch_bounds__(256)
void gemm_wmma_bf16(const unsigned short* __restrict__ A,   // M x K   row-major bf16
                    const unsigned short* __restrict__ Bm,  // K x Nc  row-major bf16
                    void* __restrict__ C,                   // M x Nc  (bf16 or f32)
                    const float* __restrict__ bias,         // Nc (nullable, f32 out only)
                    int M, int K, int Nc)
{
    constexpr int BM = 128, BN = 128, BK = 32;
    constexpr int LDA = 40;    // halves/row: 80B stride (16B-aligned, conflict-free reads)
    constexpr int LDB = 136;   // halves/row: 272B stride (16B-aligned, 2-way on K-lanes)

    __shared__ unsigned short As[2][BM * LDA];
    __shared__ unsigned short Bs[2][BK * LDB];

    const int tid   = threadIdx.x;
    const int lane  = tid & 31;
    const int wid   = tid >> 5;
    const int waveM = wid & 3;      // 0..3 -> 32 rows each
    const int waveN = wid >> 2;     // 0..1 -> 64 cols each
    const int bm    = blockIdx.y * BM;
    const int bn    = blockIdx.x * BN;

    // global->LDS staging map (16 halves = 32B per thread per tile = 2 x b128)
    const int arow = tid >> 1;            // 0..127
    const int acol = (tid & 1) * 16;      // 0 / 16
    const int brow = tid >> 3;            // 0..31
    const int bcol = (tid & 7) * 16;      // 0..112

    int gra = bm + arow;
    if (gra >= M) gra = M - 1;            // clamp: stores of these rows are guarded

    v8f acc[2][4];
    v8f zero = {0.f, 0.f, 0.f, 0.f, 0.f, 0.f, 0.f, 0.f};
    #pragma unroll
    for (int i = 0; i < 2; ++i)
        #pragma unroll
        for (int j = 0; j < 4; ++j) acc[i][j] = zero;

    auto issue = [&](int kt, int buf) {
        // A tile: 128 x 32
        const uint64_t ga =
            (uint64_t)(uintptr_t)(A + (size_t)gra * K + (size_t)kt * BK + acol);
        const uint32_t la = (uint32_t)(uintptr_t)(As[buf] + arow * LDA + acol);
        async_copy_b128(la,      ga);
        async_copy_b128(la + 16, ga + 16);
        // B tile: 32 x 128
        const uint64_t gb =
            (uint64_t)(uintptr_t)(Bm + (size_t)(kt * BK + brow) * Nc + bn + bcol);
        const uint32_t lb = (uint32_t)(uintptr_t)(Bs[buf] + brow * LDB + bcol);
        async_copy_b128(lb,      gb);
        async_copy_b128(lb + 16, gb + 16);
    };

    auto compute = [&](int buf) {
        Frag af[2], bf_[4];
        const int kh = lane >> 4;
        // A fragments: 16x32 bf16, lane(0..15)=M, lane/16 selects K half,
        // VGPR v: K = (v/4)*16 + kh*8 + (v%4)*2  (ISA 7.12.2)
        #pragma unroll
        for (int mt = 0; mt < 2; ++mt) {
            const uint32_t* rp =
                (const uint32_t*)(As[buf] + (waveM * 32 + mt * 16 + (lane & 15)) * LDA);
            #pragma unroll
            for (int v = 0; v < 8; ++v)
                af[mt].u[v] = rp[(v >> 2) * 8 + kh * 4 + (v & 3)];
        }
        // B fragments: 32x16 bf16, lane(0..15)=N, lane/16 selects K half,
        // VGPR v: K = (v/4)*16 + kh*8 + (v%4)*2 (same order as A)
        const unsigned short* rb = Bs[buf] + (lane & 15);
        #pragma unroll
        for (int nt = 0; nt < 4; ++nt) {
            const int n0 = waveN * 64 + nt * 16;
            #pragma unroll
            for (int v = 0; v < 8; ++v) {
                const int k = (v >> 2) * 16 + kh * 8 + (v & 3) * 2;
                bf_[nt].u[v] = (uint32_t)rb[k * LDB + n0] | ((uint32_t)rb[(k + 1) * LDB + n0] << 16);
            }
        }
        #pragma unroll
        for (int mt = 0; mt < 2; ++mt)
            #pragma unroll
            for (int nt = 0; nt < 4; ++nt)
                acc[mt][nt] = __builtin_amdgcn_wmma_f32_16x16x32_bf16(
                    false, af[mt].v, false, bf_[nt].v,
                    (short)0, acc[mt][nt], false, false);
    };

    const int ksteps = K / BK;
    issue(0, 0);
    for (int kt = 0; kt < ksteps; ++kt) {
        wait_async0();          // this wave's DMA for tile kt has landed in LDS
        __syncthreads();        // all waves' tile-kt data visible; buf nxt free
        if (kt + 1 < ksteps) issue(kt + 1, (kt + 1) & 1);  // DMA overlaps WMMA
        compute(kt & 1);
    }

    // Epilogue. D layout: lane(0..15)=N, lanes16-31 -> M+8; VGPR r -> M row.
    const int nn = lane & 15;
    const int mb = (lane >> 4) * 8;
    const bool full = (bm + BM) <= M;   // only the last M-block is partial
    #pragma unroll
    for (int mt = 0; mt < 2; ++mt) {
        #pragma unroll
        for (int nt = 0; nt < 4; ++nt) {
            const int gcol = bn + waveN * 64 + nt * 16 + nn;
            const int rbase = bm + waveM * 32 + mt * 16 + mb;
            if (full) {
                #pragma unroll
                for (int r = 0; r < 8; ++r) {
                    float val = acc[mt][nt][r];
                    if (OUT_BF16) {
                        ((unsigned short*)C)[(size_t)(rbase + r) * Nc + gcol] = f2bf(val);
                    } else {
                        if (bias) val += bias[gcol];
                        ((float*)C)[(size_t)(rbase + r) * Nc + gcol] = val;
                    }
                }
            } else {
                #pragma unroll
                for (int r = 0; r < 8; ++r) {
                    if (rbase + r < M) {
                        float val = acc[mt][nt][r];
                        if (OUT_BF16) {
                            ((unsigned short*)C)[(size_t)(rbase + r) * Nc + gcol] = f2bf(val);
                        } else {
                            if (bias) val += bias[gcol];
                            ((float*)C)[(size_t)(rbase + r) * Nc + gcol] = val;
                        }
                    }
                }
            }
        }
    }
}

// ---------------------------------------------------------------------------
// Criss-cross attention for the 56x56 grid tokens.
// One 64-thread block per (batch, head, column j); thread i = query (i, j).
// Column K/V (cls + 56 rows) staged in LDS; row keys streamed (L2 resident:
// the whole 116MB QKV tensor fits in the 192MB L2).
// ---------------------------------------------------------------------------
__global__ __launch_bounds__(64)
void cc_attn_kernel(const unsigned short* __restrict__ qkv,  // (B*N) x 2304 bf16
                    unsigned short* __restrict__ aout,       // (B*N) x 768  bf16
                    int B, int Hs, int Ws, int N)
{
    __shared__ float kcol[57][HD];
    __shared__ float vcol[57][HD];
    __shared__ float sc[56][116];   // 113 scores used per query

    const int bid = blockIdx.x;
    const int j = bid % Ws;
    const int h = (bid / Ws) % NHEADS;
    const int b = bid / (Ws * NHEADS);
    const int tid = threadIdx.x;

    // cooperative stage of column keys/values: slot 0 = cls, slot r+1 = (r, j)
    for (int s = 0; s < 1 + Hs; ++s) {
        const int tok = (s == 0) ? 0 : (1 + (s - 1) * Ws + j);
        const size_t base = (size_t)(b * N + tok) * QKVDIM + h * HD;
        kcol[s][tid] = bf2f(qkv[base + CDIM + tid]);
        vcol[s][tid] = bf2f(qkv[base + 2 * CDIM + tid]);
    }
    __syncthreads();

    if (tid < Hs) {
        const int i = tid;
        const int qtok = 1 + i * Ws + j;
        const size_t qbase = (size_t)(b * N + qtok) * QKVDIM + h * HD;

        float q[HD];
        #pragma unroll
        for (int d = 0; d < HD; ++d) q[d] = bf2f(qkv[qbase + d]) * 0.125f; // 1/sqrt(64)

        float mx = -3.4e38f;
        // column scores (57)
        for (int s = 0; s < 1 + Hs; ++s) {
            float a = 0.f;
            #pragma unroll
            for (int d = 0; d < HD; ++d) a += q[d] * kcol[s][d];
            sc[i][s] = a;
            mx = fmaxf(mx, a);
        }
        // row scores (56, diagonal c==j masked out)
        for (int c = 0; c < Ws; ++c) {
            if (c == j) continue;
            const int kt = 1 + i * Ws + c;
            const unsigned short* kp =
                qkv + (size_t)(b * N + kt) * QKVDIM + CDIM + h * HD;
            __builtin_prefetch(kp + QKVDIM, 0, 1);
            float a = 0.f;
            #pragma unroll
            for (int d = 0; d < HD; ++d) a += q[d] * bf2f(kp[d]);
            sc[i][57 + c] = a;
            mx = fmaxf(mx, a);
        }

        float sum = 0.f;
        float o[HD];
        #pragma unroll
        for (int d = 0; d < HD; ++d) o[d] = 0.f;

        for (int s = 0; s < 1 + Hs; ++s) {
            const float w = __expf(sc[i][s] - mx);
            sum += w;
            #pragma unroll
            for (int d = 0; d < HD; ++d) o[d] += w * vcol[s][d];
        }
        for (int c = 0; c < Ws; ++c) {
            if (c == j) continue;
            const float w = __expf(sc[i][57 + c] - mx);
            sum += w;
            const int kt = 1 + i * Ws + c;
            const unsigned short* vp =
                qkv + (size_t)(b * N + kt) * QKVDIM + 2 * CDIM + h * HD;
            #pragma unroll
            for (int d = 0; d < HD; ++d) o[d] += w * bf2f(vp[d]);
        }
        const float inv = 1.f / sum;
        unsigned short* op = aout + (size_t)(b * N + qtok) * CDIM + h * HD;
        #pragma unroll
        for (int d = 0; d < HD; ++d) op[d] = f2bf(o[d] * inv);
    }
}

// ---------------------------------------------------------------------------
// CLS-token attention: q_cls attends over all N keys. One block per (b, head).
// ---------------------------------------------------------------------------
__global__ __launch_bounds__(256)
void cls_attn_kernel(const unsigned short* __restrict__ qkv,
                     unsigned short* __restrict__ aout,
                     int B, int N)
{
    __shared__ float sc[3137];
    __shared__ float qs[HD];
    __shared__ float red[256];
    __shared__ float part[4][HD];

    const int b = blockIdx.x / NHEADS;
    const int h = blockIdx.x % NHEADS;
    const int tid = threadIdx.x;
    const size_t row0 = (size_t)(b * N) * QKVDIM;

    if (tid < HD) qs[tid] = bf2f(qkv[row0 + h * HD + tid]) * 0.125f;
    __syncthreads();

    float lmx = -3.4e38f;
    for (int n = tid; n < N; n += 256) {
        const unsigned short* kp = qkv + (size_t)(b * N + n) * QKVDIM + CDIM + h * HD;
        float a = 0.f;
        #pragma unroll 8
        for (int d = 0; d < HD; ++d) a += qs[d] * bf2f(kp[d]);
        sc[n] = a;
        lmx = fmaxf(lmx, a);
    }
    red[tid] = lmx; __syncthreads();
    for (int s = 128; s > 0; s >>= 1) {
        if (tid < s) red[tid] = fmaxf(red[tid], red[tid + s]);
        __syncthreads();
    }
    const float mx = red[0];
    __syncthreads();

    float lsum = 0.f;
    for (int n = tid; n < N; n += 256) {
        const float e = __expf(sc[n] - mx);
        sc[n] = e;
        lsum += e;
    }
    red[tid] = lsum; __syncthreads();
    for (int s = 128; s > 0; s >>= 1) {
        if (tid < s) red[tid] += red[tid + s];
        __syncthreads();
    }
    const float tot = red[0];
    __syncthreads();

    const int d = tid & 63, chunk = tid >> 6;
    float a = 0.f;
    for (int n = chunk; n < N; n += 4)
        a += sc[n] * bf2f(qkv[(size_t)(b * N + n) * QKVDIM + 2 * CDIM + h * HD + d]);
    part[chunk][d] = a;
    __syncthreads();

    if (tid < HD) {
        const float o = (part[0][tid] + part[1][tid] + part[2][tid] + part[3][tid]) / tot;
        aout[(size_t)(b * N) * CDIM + h * HD + tid] = f2bf(o);
    }
}

// ---------------------------------------------------------------------------
// Host launch
// ---------------------------------------------------------------------------
extern "C" void kernel_launch(void* const* d_in, const int* in_sizes, int n_in,
                              void* d_out, int out_size, void* d_ws, size_t ws_size,
                              hipStream_t stream) {
    const float* x     = (const float*)d_in[0];   // (8, 3137, 768)
    const float* Wqkv  = (const float*)d_in[1];   // (768, 2304)
    const float* Wproj = (const float*)d_in[2];   // (768, 768)
    const float* bproj = (const float*)d_in[3];   // (768,)
    float* out = (float*)d_out;                   // (8, 3137, 768) fp32

    const int B = 8, Hs = 56, Ws = 56;
    const int N = Hs * Ws + 1;     // 3137
    const int M = B * N;           // 25096

    char* ws = (char*)d_ws;
    size_t off = 0;
    auto take = [&](size_t bytes) {
        char* p = ws + off;
        off = (off + bytes + 255) & ~(size_t)255;
        return p;
    };
    unsigned short* Xbf     = (unsigned short*)take((size_t)M * CDIM * 2);
    unsigned short* Wqkvbf  = (unsigned short*)take((size_t)CDIM * QKVDIM * 2);
    unsigned short* Wprojbf = (unsigned short*)take((size_t)CDIM * CDIM * 2);
    unsigned short* QKVbf   = (unsigned short*)take((size_t)M * QKVDIM * 2);
    unsigned short* Abf     = (unsigned short*)take((size_t)M * CDIM * 2);

    auto cdiv = [](int a, int b) { return (a + b - 1) / b; };

    const int nx = M * CDIM;
    f32_to_bf16_kernel<<<cdiv(nx, 256), 256, 0, stream>>>(x, Xbf, nx);
    f32_to_bf16_kernel<<<cdiv(CDIM * QKVDIM, 256), 256, 0, stream>>>(Wqkv, Wqkvbf, CDIM * QKVDIM);
    f32_to_bf16_kernel<<<cdiv(CDIM * CDIM, 256), 256, 0, stream>>>(Wproj, Wprojbf, CDIM * CDIM);

    // QKV = X @ W_qkv   (bf16 out, f32 accumulate in WMMA)
    dim3 g1(QKVDIM / 128, cdiv(M, 128));
    gemm_wmma_bf16<true><<<g1, 256, 0, stream>>>(Xbf, Wqkvbf, (void*)QKVbf, nullptr,
                                                 M, CDIM, QKVDIM);

    // criss-cross attention (grid tokens) + cls attention
    cc_attn_kernel<<<B * NHEADS * Ws, 64, 0, stream>>>(QKVbf, Abf, B, Hs, Ws, N);
    cls_attn_kernel<<<B * NHEADS, 256, 0, stream>>>(QKVbf, Abf, B, N);

    // out = Attn @ W_proj + b_proj   (f32 out)
    dim3 g2(CDIM / 128, cdiv(M, 128));
    gemm_wmma_bf16<false><<<g2, 256, 0, stream>>>(Abf, Wprojbf, (void*)out, bproj,
                                                  M, CDIM, CDIM);
}
